// Net_21852793602137
// MI455X (gfx1250) — matrix operand order repaired and checked by reading, not compiled, not run on hardware
//
#include <hip/hip_runtime.h>

// ---------------------------------------------------------------------------
// Graph-network (encode -> 4x process -> decode) for MI455X (gfx1250).
// Hot GEMMs run on v_wmma_f32_16x16x32_bf16 (wave32, f32 accumulate).
// All edge-GEMM operands (emb rows, h_e, x_enc, h_x, g) are stored in memory
// as bf16 rows, so A fragments are two b128 loads per K-chunk with zero VALU
// packing, and streamed h_e bytes are halved. f32 is kept for recv/sent
// atomic accumulators, the h_x skip-add state, hid, and all outputs.
// Weights staged to LDS with async-to-LDS DMA (ASYNCcnt path).
// ---------------------------------------------------------------------------

#define HSZ 64

typedef __attribute__((ext_vector_type(16))) __bf16 v16bf;
typedef __attribute__((ext_vector_type(8)))  float  v8f;
typedef __attribute__((ext_vector_type(4)))  float  v4f;
typedef __attribute__((ext_vector_type(4)))  int    v4i;
typedef unsigned int u32;
typedef unsigned short u16;

union FragU {
  v16bf bf;
  u32   u[8];
  uint4 q[2];
};

#if __has_builtin(__builtin_amdgcn_cvt_pk_bf16_f32)
typedef __bf16 bf16x2 __attribute__((ext_vector_type(2)));
__device__ __forceinline__ u32 pack2_bf16(float a, float b) {
  bf16x2 r = __builtin_amdgcn_cvt_pk_bf16_f32(a, b);
  return __builtin_bit_cast(u32, r);
}
#else
__device__ __forceinline__ u32 pack2_bf16(float a, float b) {
  u32 ua = __float_as_uint(a);
  u32 ub = __float_as_uint(b);
  ua = (ua + 0x7FFFu + ((ua >> 16) & 1u)) >> 16;       // RNE f32 -> bf16
  ub = (ub + 0x7FFFu + ((ub >> 16) & 1u)) >> 16;
  return ua | (ub << 16);
}
#endif

__device__ __forceinline__ u16 bf16_bits(float a) {
  u32 ua = __float_as_uint(a);
  return (u16)((ua + 0x7FFFu + ((ua >> 16) & 1u)) >> 16);
}

// Stage `ndw` dwords of weight fragments from global into LDS with the whole
// 256-thread block. Prefers gfx1250 async-to-LDS DMA (ASYNCcnt path).
__device__ __forceinline__ void stage_weights(u32* s_dst, const u32* g_src, int ndw) {
#if __has_builtin(__builtin_amdgcn_global_load_async_to_lds_b128)
  typedef v4i __attribute__((address_space(1)))* gv4p;   // global int4*
  typedef v4i __attribute__((address_space(3)))* lv4p;   // LDS int4*
  for (int i = (int)threadIdx.x * 4; i < ndw; i += 256 * 4) {
    gv4p g = (gv4p)(uintptr_t)(g_src + i);
    lv4p l = (lv4p)(u32)(uintptr_t)(s_dst + i);
    __builtin_amdgcn_global_load_async_to_lds_b128(g, l, 0, 0);
  }
#if __has_builtin(__builtin_amdgcn_s_wait_asynccnt)
  __builtin_amdgcn_s_wait_asynccnt(0);
#else
  asm volatile("s_wait_asynccnt 0" ::: "memory");
#endif
#else
  for (int i = (int)threadIdx.x; i < ndw; i += 256) s_dst[i] = g_src[i];
#endif
}

// Load a bf16 A fragment for K-chunk `kc` from a bf16 row (as u32*).
// Dword(v) = (kc&1)*16 + hi*4 + (v&3) + (v>>2)*8 -> two contiguous b128s.
__device__ __forceinline__ FragU load_a_bf16(const u32* row, int kc, int hi) {
  const u32* p = row + ((kc & 1) << 4) + (hi << 2);
  FragU A;
  A.q[0] = *reinterpret_cast<const uint4*>(p);
  A.q[1] = *reinterpret_cast<const uint4*>(p + 8);
  return A;
}

// ---------------------------------------------------------------------------
// Pre-swizzle a [K,64] f32 weight matrix into per-lane WMMA B fragments.
// Fragment dword index: ((kc*4 + ct)*32 + lane)*8 + v
// ---------------------------------------------------------------------------
__global__ void prep_frag_kernel(const float* __restrict__ W,
                                 u32* __restrict__ frag, int K) {
  int d = blockIdx.x * 256 + threadIdx.x;
  if (d >= K * 32) return;
  int v    = d & 7;
  int l    = (d >> 3) & 31;
  int ctkc = d >> 8;
  int ct   = ctkc & 3;
  int kc   = ctkc >> 2;
  int col  = (ct << 4) + (l & 15);
  int krow = (kc << 5) + ((l >> 4) << 3) + ((v & 3) << 1) + ((v >> 2) << 4);
  frag[d] = pack2_bf16(W[krow * HSZ + col], W[(krow + 1) * HSZ + col]);
}

// Generic f32 -> packed bf16 pair conversion (emb table, g init)
__global__ void cvt_bf16_kernel(const float* __restrict__ src,
                                u32* __restrict__ dst, int npairs) {
  int d = blockIdx.x * 256 + threadIdx.x;
  if (d >= npairs) return;
  dst[d] = pack2_bf16(src[2 * d], src[2 * d + 1]);
}

// x_enc = relu(node_attr @ W_ne + b_ne), IN_F = 4; emitted directly as bf16
__global__ void encode_nodes_kernel(const float* __restrict__ node_attr,
                                    const float* __restrict__ W_ne,
                                    const float* __restrict__ b_ne,
                                    u32* __restrict__ x_bf, int TN) {
  int idx = blockIdx.x * 256 + threadIdx.x;                 // over TN*32 pairs
  if (idx >= TN * 32) return;
  int row = idx >> 5, c = (idx & 31) << 1;
  const float4 na = *reinterpret_cast<const float4*>(node_attr + (size_t)row * 4);
  float a0 = b_ne[c], a1 = b_ne[c + 1];
  a0 += na.x * W_ne[0 * HSZ + c];     a1 += na.x * W_ne[0 * HSZ + c + 1];
  a0 += na.y * W_ne[1 * HSZ + c];     a1 += na.y * W_ne[1 * HSZ + c + 1];
  a0 += na.z * W_ne[2 * HSZ + c];     a1 += na.z * W_ne[2 * HSZ + c + 1];
  a0 += na.w * W_ne[3 * HSZ + c];     a1 += na.w * W_ne[3 * HSZ + c + 1];
  x_bf[idx] = pack2_bf16(fmaxf(a0, 0.f), fmaxf(a1, 0.f));
}

// ---------------------------------------------------------------------------
// Edge block: e_new = relu([e_t, h_e, x_s, hx_s, x_r, hx_r, g] @ W_eb + b_eb)
// One wave per 16-edge tile; all A operands bf16 in memory -> 2 b128 loads
// per chunk, no packing. 14 chunks x 4 col-tiles = 56 WMMAs per tile.
// ---------------------------------------------------------------------------
__global__ void __launch_bounds__(256)
edge_block_kernel(const u32*   __restrict__ emb_bf,   // [NUM_EMB][32] dwords
                  const int*   __restrict__ edge_attr_t,
                  const int*   __restrict__ edge_index,
                  const u32*   __restrict__ x_bf,     // [N][32]
                  const u32*   __restrict__ hx_bf,    // [N][32]
                  u32*         __restrict__ h_e,      // [E][32] in/out (bf16)
                  const u32*   __restrict__ g_bf,     // [32]
                  const float* __restrict__ b_eb,
                  const u32*   __restrict__ w_frag,   // 448*32 dwords
                  float* __restrict__ recv,
                  float* __restrict__ sent,
                  float* __restrict__ e_sum,
                  int E) {
  __shared__ __align__(16) u32 s_w[448 * 32];          // 56 KB bf16 weights
  __shared__ float s_bias[HSZ];
  __shared__ float s_csum[HSZ];

  stage_weights(s_w, w_frag, 448 * 32);
  if (threadIdx.x < HSZ) { s_bias[threadIdx.x] = b_eb[threadIdx.x]; s_csum[threadIdx.x] = 0.f; }
  __syncthreads();

  const int lane  = threadIdx.x & 31;
  const int wave  = threadIdx.x >> 5;
  const int lrow  = lane & 15;
  const int hi    = lane >> 4;
  const int hiOff = hi << 3;
  const int tileBase = (blockIdx.x * 8 + wave) << 4;

  if (tileBase < E) {
    int e = tileBase + lrow;
    if (e >= E) e = E - 1;
    const int sN = edge_index[e];
    const int rN = edge_index[E + e];
    const int aI = edge_attr_t[e];

    const u32* seg[7];
    seg[0] = emb_bf + (size_t)aI * 32;
    seg[1] = h_e    + (size_t)e  * 32;
    seg[2] = x_bf   + (size_t)sN * 32;
    seg[3] = hx_bf  + (size_t)sN * 32;
    seg[4] = x_bf   + (size_t)rN * 32;
    seg[5] = hx_bf  + (size_t)rN * 32;
    seg[6] = g_bf;

    v8f acc[4] = {};
#pragma unroll
    for (int kc = 0; kc < 14; ++kc) {
      FragU A = load_a_bf16(seg[kc >> 1], kc, hi);
#pragma unroll
      for (int ct = 0; ct < 4; ++ct) {
        FragU B;
        const uint4* bp =
            reinterpret_cast<const uint4*>(&s_w[((((kc << 2) | ct) << 5) + lane) << 3]);
        B.q[0] = bp[0];
        B.q[1] = bp[1];
        acc[ct] = __builtin_amdgcn_wmma_f32_16x16x32_bf16(
            false, A.bf, false, B.bf, (short)0, acc[ct], false, false);
      }
    }

    // C/D layout: lane -> column (ct*16 + lane&15), vgpr j -> row hiOff+j.
    u16* h_e16 = reinterpret_cast<u16*>(h_e);
#pragma unroll
    for (int ct = 0; ct < 4; ++ct) {
      const int n = (ct << 4) + lrow;
      const float bias = s_bias[n];
      float csum = 0.f;
#pragma unroll
      for (int j = 0; j < 8; ++j) {
        const int eo = tileBase + hiOff + j;
        if (eo < E) {
          float val = fmaxf(acc[ct][j] + bias, 0.f);
          h_e16[(size_t)eo * HSZ + n] = bf16_bits(val);
          const int so = edge_index[eo];
          const int ro = edge_index[E + eo];
          atomicAdd(&recv[(size_t)ro * HSZ + n], val);
          atomicAdd(&sent[(size_t)so * HSZ + n], val);
          csum += val;
        }
      }
      atomicAdd(&s_csum[n], csum);                     // ds_add_f32
    }
  }
  __syncthreads();
  if (threadIdx.x < HSZ) atomicAdd(&e_sum[threadIdx.x], s_csum[threadIdx.x]);
}

// ---------------------------------------------------------------------------
// Node block: x_new = relu([recv, sent, x_t, h_x, g] @ W_nb + b_nb)
// recv/sent come from f32 atomics (pack at load); x_t/h_x/g are bf16 rows.
// h_x += x_new (skip), hid_t = h_x_next (f32), hx_bf mirror, time_t = x_new.
// ---------------------------------------------------------------------------
__global__ void __launch_bounds__(256)
node_block_kernel(const float* __restrict__ recv,
                  const float* __restrict__ sent,
                  const u32*   __restrict__ x_bf,
                  float*       __restrict__ h_x,      // f32 state
                  u32*         __restrict__ hx_bf,    // bf16 mirror
                  const u32*   __restrict__ g_bf,
                  const float* __restrict__ b_nb,
                  const u32*   __restrict__ w_frag,   // 320*32 dwords
                  float* __restrict__ hid_t,
                  float* __restrict__ time_t,
                  float* __restrict__ x_sum,
                  int N) {
  __shared__ __align__(16) u32 s_w[320 * 32];          // 40 KB
  __shared__ float s_bias[HSZ];
  __shared__ float s_csum[HSZ];

  stage_weights(s_w, w_frag, 320 * 32);
  if (threadIdx.x < HSZ) { s_bias[threadIdx.x] = b_nb[threadIdx.x]; s_csum[threadIdx.x] = 0.f; }
  __syncthreads();

  const int lane  = threadIdx.x & 31;
  const int wave  = threadIdx.x >> 5;
  const int lrow  = lane & 15;
  const int hi    = lane >> 4;
  const int hiOff = hi << 3;
  const int tileBase = (blockIdx.x * 8 + wave) << 4;

  if (tileBase < N) {
    int nd = tileBase + lrow;
    if (nd >= N) nd = N - 1;

    const float* fseg[2];
    fseg[0] = recv + (size_t)nd * HSZ;
    fseg[1] = sent + (size_t)nd * HSZ;
    const u32* bseg[3];
    bseg[0] = x_bf  + (size_t)nd * 32;
    bseg[1] = hx_bf + (size_t)nd * 32;
    bseg[2] = g_bf;

    v8f acc[4] = {};
#pragma unroll
    for (int kc = 0; kc < 10; ++kc) {
      FragU A;
      if (kc < 4) {                                    // f32 accumulators
        const float* p = fseg[kc >> 1] + ((kc & 1) << 5);
#pragma unroll
        for (int v = 0; v < 8; ++v) {
          const int o = hiOff + ((v & 3) << 1) + ((v >> 2) << 4);
          const float2 f = *reinterpret_cast<const float2*>(p + o);
          A.u[v] = pack2_bf16(f.x, f.y);
        }
      } else {                                         // bf16 rows, direct
        A = load_a_bf16(bseg[(kc - 4) >> 1], kc, hi);
      }
#pragma unroll
      for (int ct = 0; ct < 4; ++ct) {
        FragU B;
        const uint4* bp =
            reinterpret_cast<const uint4*>(&s_w[((((kc << 2) | ct) << 5) + lane) << 3]);
        B.q[0] = bp[0];
        B.q[1] = bp[1];
        acc[ct] = __builtin_amdgcn_wmma_f32_16x16x32_bf16(
            false, A.bf, false, B.bf, (short)0, acc[ct], false, false);
      }
    }

    u16* hx16 = reinterpret_cast<u16*>(hx_bf);
#pragma unroll
    for (int ct = 0; ct < 4; ++ct) {
      const int n = (ct << 4) + lrow;
      const float bias = s_bias[n];
      float csum = 0.f;
#pragma unroll
      for (int j = 0; j < 8; ++j) {
        const int no = tileBase + hiOff + j;
        if (no < N) {
          float xn = fmaxf(acc[ct][j] + bias, 0.f);
          float hv = h_x[(size_t)no * HSZ + n] + xn;   // skip connection
          h_x  [(size_t)no * HSZ + n] = hv;
          hid_t[(size_t)no * HSZ + n] = hv;
          hx16 [(size_t)no * HSZ + n] = bf16_bits(hv);
          __builtin_nontemporal_store(xn, &time_t[(size_t)no * HSZ + n]);
          csum += xn;
        }
      }
      atomicAdd(&s_csum[n], csum);
    }
  }
  __syncthreads();
  if (threadIdx.x < HSZ) atomicAdd(&x_sum[threadIdx.x], s_csum[threadIdx.x]);
}

// Global block: g = relu([e_mean, x_mean, g] @ W_gb + b_gb)  (tiny; VALU)
__global__ void global_block_kernel(const float* __restrict__ e_sum,
                                    const float* __restrict__ x_sum,
                                    float* __restrict__ g_cur,
                                    u32* __restrict__ g_bf,
                                    const float* __restrict__ W_gb,
                                    const float* __restrict__ b_gb,
                                    float invE, float invN) {
  __shared__ float gin[3 * HSZ];
  const int t = threadIdx.x;                           // 64 threads
  gin[t]           = e_sum[t] * invE;
  gin[HSZ + t]     = x_sum[t] * invN;
  gin[2 * HSZ + t] = g_cur[t];
  __syncthreads();
  float acc = b_gb[t];
  for (int k = 0; k < 3 * HSZ; ++k) acc += gin[k] * W_gb[k * HSZ + t];
  acc = fmaxf(acc, 0.f);
  g_cur[t] = acc;
  reinterpret_cast<u16*>(g_bf)[t] = bf16_bits(acc);
}

// spat_d[i,:] += coeff * L_values[z] * hid[j,:]  (scatter segment-sum)
__global__ void laplacian_kernel(const float* __restrict__ L_values,
                                 const int* __restrict__ L_index,
                                 const float* __restrict__ coeff,
                                 const float* __restrict__ hid_t,
                                 float* __restrict__ spat_t,
                                 int NNZ) {
  long idx = (long)blockIdx.x * 256 + threadIdx.x;
  long z = idx >> 6;
  if (z >= NNZ) return;
  int h = (int)(idx & 63);
  int i = L_index[z];
  int j = L_index[(size_t)NNZ + z];
  float v = coeff[0] * L_values[z] * hid_t[(size_t)j * HSZ + h];
  atomicAdd(&spat_t[(size_t)i * HSZ + h], v);
}

// ---------------------------------------------------------------------------
// Decoders: out_nodes = relu(hid@W_d1+b)@W_d2+b2 ; pred_in = relu(hid@W_i1+b)@W_i2+b2
// 64x64 GEMMs on WMMA (hid is f32, packed at load); second GEMM via LDS.
// ---------------------------------------------------------------------------
__global__ void __launch_bounds__(256)
decoder_kernel(const float* __restrict__ hid,
               const u32* __restrict__ frag_d1,
               const u32* __restrict__ frag_i1,
               const float* __restrict__ b_d1,
               const float* __restrict__ b_i1,
               const float* __restrict__ W_d2, const float* __restrict__ b_d2,
               const float* __restrict__ W_i2, const float* __restrict__ b_i2,
               float* __restrict__ out_nodes,
               float* __restrict__ pred_in,
               int TN) {
  __shared__ __align__(16) u32 s_w[2][64 * 32];        // 16 KB
  __shared__ float s_b[2][HSZ];
  __shared__ float s_t1[8][16][HSZ + 1];               // 33 KB, padded

  stage_weights(&s_w[0][0], frag_d1, 64 * 32);
  stage_weights(&s_w[1][0], frag_i1, 64 * 32);
  if (threadIdx.x < HSZ) { s_b[0][threadIdx.x] = b_d1[threadIdx.x]; s_b[1][threadIdx.x] = b_i1[threadIdx.x]; }
  __syncthreads();

  const int lane  = threadIdx.x & 31;
  const int wave  = threadIdx.x >> 5;
  const int lrow  = lane & 15;
  const int hiOff = (lane >> 4) << 3;
  const int tileBase = (blockIdx.x * 8 + wave) << 4;

  int row = tileBase + lrow;
  if (row >= TN) row = TN - 1;
  const float* p0 = hid + (size_t)row * HSZ;

  for (int dec = 0; dec < 2; ++dec) {
    if (tileBase < TN) {
      v8f acc[4] = {};
#pragma unroll
      for (int kc = 0; kc < 2; ++kc) {
        const float* p = p0 + (kc << 5);
        FragU A;
#pragma unroll
        for (int v = 0; v < 8; ++v) {
          const int o = hiOff + ((v & 3) << 1) + ((v >> 2) << 4);
          const float2 f = *reinterpret_cast<const float2*>(p + o);
          A.u[v] = pack2_bf16(f.x, f.y);
        }
#pragma unroll
        for (int ct = 0; ct < 4; ++ct) {
          FragU B;
          const uint4* bp =
              reinterpret_cast<const uint4*>(&s_w[dec][((((kc << 2) | ct) << 5) + lane) << 3]);
          B.q[0] = bp[0];
          B.q[1] = bp[1];
          acc[ct] = __builtin_amdgcn_wmma_f32_16x16x32_bf16(
              false, A.bf, false, B.bf, (short)0, acc[ct], false, false);
        }
      }
#pragma unroll
      for (int ct = 0; ct < 4; ++ct) {
        const int n = (ct << 4) + lrow;
        const float bias = s_b[dec][n];
#pragma unroll
        for (int j = 0; j < 8; ++j)
          s_t1[wave][hiOff + j][n] = fmaxf(acc[ct][j] + bias, 0.f);
      }
    }
    __syncthreads();
    if (tileBase < TN && lane < 16) {
      const int rowG = tileBase + lane;
      if (rowG < TN) {
        const float* t1 = s_t1[wave][lane];
        if (dec == 0) {
          float o = b_d2[0];
          for (int c = 0; c < HSZ; ++c) o += t1[c] * W_d2[c];
          __builtin_nontemporal_store(o, &out_nodes[rowG]);
        } else {
          float o0 = b_i2[0], o1 = b_i2[1], o2 = b_i2[2], o3 = b_i2[3];
          for (int c = 0; c < HSZ; ++c) {
            const float tv = t1[c];
            o0 += tv * W_i2[c * 4 + 0];
            o1 += tv * W_i2[c * 4 + 1];
            o2 += tv * W_i2[c * 4 + 2];
            o3 += tv * W_i2[c * 4 + 3];
          }
          v4f o4 = {o0, o1, o2, o3};
          __builtin_nontemporal_store(o4, reinterpret_cast<v4f*>(&pred_in[(size_t)rowG * 4]));
        }
      }
    }
    __syncthreads();
  }
}

// ---------------------------------------------------------------------------
extern "C" void kernel_launch(void* const* d_in, const int* in_sizes, int n_in,
                              void* d_out, int out_size, void* d_ws, size_t ws_size,
                              hipStream_t stream) {
  (void)n_in; (void)out_size; (void)ws_size;

  const float* node_attr   = (const float*)d_in[0];
  const float* global_attr = (const float*)d_in[1];
  const float* L_values    = (const float*)d_in[2];
  const float* coeff       = (const float*)d_in[3];
  const int*   edge_attr   = (const int*)d_in[4];
  const int*   edge_index  = (const int*)d_in[5];
  const int*   L_index     = (const int*)d_in[6];
  const float* emb         = (const float*)d_in[8];
  const float* W_ne = (const float*)d_in[9],  * b_ne = (const float*)d_in[10];
  const float* W_eb = (const float*)d_in[11], * b_eb = (const float*)d_in[12];
  const float* W_nb = (const float*)d_in[13], * b_nb = (const float*)d_in[14];
  const float* W_gb = (const float*)d_in[15], * b_gb = (const float*)d_in[16];
  const float* W_d1 = (const float*)d_in[17], * b_d1 = (const float*)d_in[18];
  const float* W_d2 = (const float*)d_in[19], * b_d2 = (const float*)d_in[20];
  const float* W_i1 = (const float*)d_in[21], * b_i1 = (const float*)d_in[22];
  const float* W_i2 = (const float*)d_in[23], * b_i2 = (const float*)d_in[24];

  const int NNZ = in_sizes[2];
  const int E   = in_sizes[5] / 2;
  const int T   = in_sizes[4] / E;
  const int N   = in_sizes[0] / (T * 4);
  const int TN  = T * N;
  const int NE  = in_sizes[8] / HSZ;                   // NUM_EMB

  // Output tuple layout (flat, in return order)
  float* out_nodes = (float*)d_out;                    // [T,N,1]
  float* time_d    = out_nodes + (size_t)TN;           // [T,N,64]
  float* spat_d    = time_d + (size_t)TN * HSZ;        // [T,N,64]
  float* pred_in   = spat_d + (size_t)TN * HSZ;        // [T,N,4]

  // Workspace carve (~92 MB for T=4,N=20000,E=320000)
  float* wp = (float*)d_ws;
  float* hid   = wp; wp += (size_t)TN * HSZ;           // f32 carried state
  float* h_x   = wp; wp += (size_t)N * HSZ;            // f32 skip-add state
  u32*  hx_bf  = (u32*)wp; wp += (size_t)N * 32;       // bf16 mirror (adjacent: one memset)
  float* recv  = wp; wp += (size_t)N * HSZ;            // recv..x_sum contiguous
  float* sent  = wp; wp += (size_t)N * HSZ;            //   (single memset zone)
  float* e_sum = wp; wp += HSZ;
  float* x_sum = wp; wp += HSZ;
  float* g_cur = wp; wp += HSZ;
  u32*  g_bf   = (u32*)wp; wp += 32;
  u32*  x_bf   = (u32*)wp; wp += (size_t)TN * 32;      // x_enc in bf16
  u32*  h_e    = (u32*)wp; wp += (size_t)E * 32;       // h_e in bf16
  u32*  emb_bf = (u32*)wp; wp += (size_t)NE * 32;
  u32* frag_eb = (u32*)wp; wp += 448 * 32;
  u32* frag_nb = (u32*)wp; wp += 320 * 32;
  u32* frag_d1 = (u32*)wp; wp += 64 * 32;
  u32* frag_i1 = (u32*)wp; wp += 64 * 32;

  // Weight pre-swizzle (bf16 WMMA B fragments) + operand conversions
  prep_frag_kernel<<<(448 * 32 + 255) / 256, 256, 0, stream>>>(W_eb, frag_eb, 448);
  prep_frag_kernel<<<(320 * 32 + 255) / 256, 256, 0, stream>>>(W_nb, frag_nb, 320);
  prep_frag_kernel<<<(64 * 32 + 255) / 256, 256, 0, stream>>>(W_d1, frag_d1, 64);
  prep_frag_kernel<<<(64 * 32 + 255) / 256, 256, 0, stream>>>(W_i1, frag_i1, 64);
  cvt_bf16_kernel<<<(NE * 32 + 255) / 256, 256, 0, stream>>>(emb, emb_bf, NE * 32);
  cvt_bf16_kernel<<<1, 64, 0, stream>>>(global_attr, g_bf, 32);

  // Initial state: h_x = 0 (f32 + bf16 mirror, contiguous), h_e = 0, spat_d = 0
  (void)hipMemsetAsync(h_x, 0, ((size_t)N * HSZ + (size_t)N * 32) * sizeof(float), stream);
  (void)hipMemsetAsync(h_e, 0, (size_t)E * 32 * sizeof(u32), stream);
  (void)hipMemsetAsync(spat_d, 0, (size_t)TN * HSZ * sizeof(float), stream);
  (void)hipMemcpyAsync(g_cur, global_attr, HSZ * sizeof(float), hipMemcpyDeviceToDevice, stream);

  encode_nodes_kernel<<<(TN * 32 + 255) / 256, 256, 0, stream>>>(node_attr, W_ne, b_ne, x_bf, TN);

  for (int t = 0; t < T; ++t) {
    // zero recv, sent, e_sum, x_sum in one shot
    (void)hipMemsetAsync(recv, 0, ((size_t)2 * N * HSZ + 2 * HSZ) * sizeof(float), stream);

    edge_block_kernel<<<(E + 127) / 128, 256, 0, stream>>>(
        emb_bf, edge_attr + (size_t)t * E, edge_index,
        x_bf + (size_t)t * N * 32, hx_bf, h_e, g_bf, b_eb, frag_eb,
        recv, sent, e_sum, E);

    node_block_kernel<<<(N + 127) / 128, 256, 0, stream>>>(
        recv, sent, x_bf + (size_t)t * N * 32, h_x, hx_bf, g_bf, b_nb, frag_nb,
        hid + (size_t)t * N * HSZ, time_d + (size_t)t * N * HSZ, x_sum, N);

    global_block_kernel<<<1, 64, 0, stream>>>(e_sum, x_sum, g_cur, g_bf, W_gb, b_gb,
                                              1.0f / (float)E, 1.0f / (float)N);

    laplacian_kernel<<<(unsigned)(((long)NNZ * HSZ + 255) / 256), 256, 0, stream>>>(
        L_values, L_index, coeff,
        hid + (size_t)t * N * HSZ, spat_d + (size_t)t * N * HSZ, NNZ);
  }

  decoder_kernel<<<(TN + 127) / 128, 256, 0, stream>>>(
      hid, frag_d1, frag_i1, b_d1, b_i1, W_d2, b_d2, W_i2, b_i2,
      out_nodes, pred_in, TN);
}
